// UniPhyBlock_88081189306895
// MI455X (gfx1250) — matrix operand order repaired and checked by reading, not compiled, not run on hardware
//
#include <hip/hip_runtime.h>

// ---------------------------------------------------------------------------
// UniPhyBlock for MI455X (gfx1250, wave32). All dense GEMMs run on
// v_wmma_f32_16x16x32_f16 (f16 in, f32 accumulate). Complex matmuls are
// packed as real GEMMs with doubled K. Scan / LayerNorm / softmax / flux are
// scalar fp32 (bandwidth-trivial).
// ---------------------------------------------------------------------------

typedef __attribute__((ext_vector_type(16))) _Float16 v16h;
typedef __attribute__((ext_vector_type(8)))  _Float16 v8h;
typedef __attribute__((ext_vector_type(8)))  float    v8f;

#define WMMA(a,b,c) __builtin_amdgcn_wmma_f32_16x16x32_f16(false,(a),false,(b),(short)0,(c),false,false)

// Problem sizes
#define TT   16
#define DD   64
#define HWN  1024            // H*W
#define MTOT 16384           // T*H*W
#define EE   8
#define HIDC 256

// Output offsets (floats)
#define OFF_H    2097152     // 2*T*D*H*W
#define OFF_FLUX 2228224     // OFF_H + 2*HW*D

// Workspace layout (bytes) ---------------------------------------------------
static constexpr size_t WS_WENC = 0;                                  // f16 [128][128]
static constexpr size_t WS_WDEC = WS_WENC + 128*128*2;
static constexpr size_t WS_W1T  = WS_WDEC + 128*128*2;                // f16 [8][512][128]
static constexpr size_t WS_W2P  = WS_W1T  + (size_t)8*512*128*2;      // f16 [8][16][128][32]
static constexpr size_t WS_B1   = WS_W2P  + (size_t)8*128*512*2;      // f32 [8][512]
static constexpr size_t WS_B2   = WS_B1   + 8*512*4;                  // f32 [8][128]
static constexpr size_t WS_XCAT = WS_B2   + 8*128*4;                  // f16 [16384][128]
static constexpr size_t WS_XEIG = WS_XCAT + (size_t)MTOT*128*2;       // f32 [16384][128]
static constexpr size_t WS_FLUX = WS_XEIG + (size_t)MTOT*128*4;       // f32 [16][64]
static constexpr size_t WS_ATAB = WS_FLUX + 1024*4;                   // f32 [16][64][2]
static constexpr size_t WS_FTAB = WS_ATAB + 2048*4;                   // f32 [16][64][2]
static constexpr size_t WS_UCAT = WS_FTAB + 2048*4;                   // f16 [16384][128]
static constexpr size_t WS_XC   = WS_UCAT + (size_t)MTOT*128*2;       // f32 [16384][128]
static constexpr size_t WS_XC16 = WS_XC   + (size_t)MTOT*128*4;       // f16 [16384][128]
static constexpr size_t WS_P    = WS_XC16 + (size_t)MTOT*128*2;       // f32 [16384][8]

// ---------------------------------------------------------------------------
// Fragment loaders (ISA 7.12.2, wave32, 16-bit operands, K=32)
// A: lane L -> row M = L%16 ; element j -> K = j + 8*g (j<8), j+8+8*g (j>=8)
// B: stored [N][K]; lane L -> col N = L%16 ; element j -> K = j + 8*g (j<8), j+8+8*g (j>=8)
__device__ __forceinline__ v16h load_fragA(const _Float16* rowp, int g) {
  v8h lo = *(const v8h*)(rowp + 8*g);
  v8h hi = *(const v8h*)(rowp + 16 + 8*g);
  v16h f;
#pragma unroll
  for (int i = 0; i < 8; ++i) { f[i] = lo[i]; f[i+8] = hi[i]; }
  return f;
}
__device__ __forceinline__ v16h load_fragB(const _Float16* rowp, int g) {
  v8h lo = *(const v8h*)(rowp + 8*g);
  v8h hi = *(const v8h*)(rowp + 16 + 8*g);
  v16h f;
#pragma unroll
  for (int i = 0; i < 8; ++i) { f[i] = lo[i]; f[i+8] = hi[i]; }
  return f;
}

__device__ __forceinline__ float gelu_tanh(float x) {
  float x3 = x*x*x;
  return 0.5f*x*(1.0f + tanhf(0.7978845608028654f*(x + 0.044715f*x3)));
}

// ---------------------------------------------------------------------------
// Packing kernels
// ---------------------------------------------------------------------------
// Complex (64x64) -> real 128x128, stored transposed [n][k].
__global__ void k_pack_encdec(const float* __restrict__ Ure, const float* __restrict__ Uim,
                              _Float16* __restrict__ Wt) {
  int i = blockIdx.x*256 + threadIdx.x;
  if (i >= 128*128) return;
  int n = i >> 7, k = i & 127;
  int kr = k & 63, nr = n & 63;
  float v;
  if (n < 64) v = (k < 64) ? Ure[kr*64+nr] : -Uim[kr*64+nr];
  else        v = (k < 64) ? Uim[kr*64+nr] :  Ure[kr*64+nr];
  Wt[n*128 + k] = (_Float16)v;
}

// W1 (E,64,256) complex -> real 128x512 per expert, stored [e][n(512)][k(128)]
__global__ void k_pack_w1(const float* __restrict__ W1re, const float* __restrict__ W1im,
                          _Float16* __restrict__ W1T) {
  int i = blockIdx.x*256 + threadIdx.x;
  if (i >= 8*512*128) return;
  int e = i >> 16, rem = i & 65535;
  int n = rem >> 7, k = rem & 127;
  int kr = k & 63, nr = n & 255;
  float v;
  if (n < 256) v = (k < 64) ? W1re[(e*64+kr)*256+nr] : -W1im[(e*64+kr)*256+nr];
  else         v = (k < 64) ? W1im[(e*64+kr)*256+nr] :  W1re[(e*64+kr)*256+nr];
  W1T[((size_t)e*512 + n)*128 + k] = (_Float16)v;
}

// W2 (E,256,64) complex -> real 512x128 per expert, chunk-packed [e][kc][n(128)][kl(32)]
__global__ void k_pack_w2(const float* __restrict__ W2re, const float* __restrict__ W2im,
                          _Float16* __restrict__ W2P) {
  int i = blockIdx.x*256 + threadIdx.x;
  if (i >= 8*128*512) return;
  int e = i >> 16, rem = i & 65535;
  int n = rem >> 9, k = rem & 511;
  int kr = k & 255, nr = n & 63;
  float v;
  if (n < 64) v = (k < 256) ? W2re[(e*256+kr)*64+nr] : -W2im[(e*256+kr)*64+nr];
  else        v = (k < 256) ? W2im[(e*256+kr)*64+nr] :  W2re[(e*256+kr)*64+nr];
  int kc = k >> 5, kl = k & 31;
  W2P[(((size_t)e*16 + kc)*128 + n)*32 + kl] = (_Float16)v;
}

// biases + flux accumulator zero
__global__ void k_pack_misc(const float* __restrict__ b1re, const float* __restrict__ b1im,
                            const float* __restrict__ b2re, const float* __restrict__ b2im,
                            float* __restrict__ b1cat, float* __restrict__ b2cat,
                            float* __restrict__ fluxacc) {
  int i = blockIdx.x*256 + threadIdx.x;
  if (i < 4096) {
    int e = i >> 9, n = i & 511;
    b1cat[i] = (n < 256) ? b1re[e*256+n] : b1im[e*256+n-256];
  } else if (i < 5120) {
    int j = i - 4096, e = j >> 7, n = j & 127;
    b2cat[j] = (n < 64) ? b2re[e*64+n] : b2im[e*64+n-64];
  } else if (i < 6144) {
    fluxacc[i - 5120] = 0.0f;
  }
}

// x (1,T,D,H,W) re/im -> token-major f16 cat: Xcat[m][k], k<64 re, k>=64 im
__global__ void k_pack_x(const float* __restrict__ xre, const float* __restrict__ xim,
                         _Float16* __restrict__ Xcat) {
  int i = blockIdx.x*256 + threadIdx.x;
  if (i >= MTOT*128) return;
  int m = i >> 7, k = i & 127;
  int t = m >> 10, hw = m & 1023, d = k & 63;
  float v = (k < 64) ? xre[(t*64+d)*1024 + hw] : xim[(t*64+d)*1024 + hw];
  Xcat[i] = (_Float16)v;
}

// ---------------------------------------------------------------------------
// Encoder: xeig = Xcat @ WencT (WMMA), plus |xeig| flux accumulation
// ---------------------------------------------------------------------------
__global__ void __launch_bounds__(256) k_enc(const _Float16* __restrict__ Xcat,
                                             const _Float16* __restrict__ WencT,
                                             float* __restrict__ xeig,
                                             float* __restrict__ fluxacc) {
  __shared__ __align__(16) _Float16 sW[128*128];
  {
    const uint4* s = (const uint4*)WencT;
    uint4* d = (uint4*)sW;
    for (int i = threadIdx.x; i < 128*128/8; i += 256) d[i] = s[i];
  }
  __syncthreads();
  int lane = threadIdx.x & 31, wave = threadIdx.x >> 5;
  int g = lane >> 4, ln = lane & 15;
  int m0 = (blockIdx.x*8 + wave) * 16;

  v16h a[4];
#pragma unroll
  for (int kt = 0; kt < 4; ++kt) a[kt] = load_fragA(Xcat + (size_t)(m0+ln)*128 + kt*32, g);

  v8f acc[8];
#pragma unroll
  for (int nt = 0; nt < 8; ++nt) {
    v8f c = {0.f,0.f,0.f,0.f,0.f,0.f,0.f,0.f};
#pragma unroll
    for (int kt = 0; kt < 4; ++kt) {
      v16h b = load_fragB(sW + (nt*16 + ln)*128 + kt*32, g);
      c = WMMA(a[kt], b, c);
    }
    acc[nt] = c;
  }
  // store x_eigen (cat layout, f32) — C layout: N = ln + 16*nt, M = r + 8*g
#pragma unroll
  for (int nt = 0; nt < 8; ++nt)
#pragma unroll
    for (int r = 0; r < 8; ++r)
      xeig[(size_t)(m0 + r + 8*g)*128 + nt*16 + ln] = acc[nt][r];

  // flux: sum_{m in tile} |xeig| per d ; re at nt, im at nt+4 (same lane/reg)
  int t = m0 >> 10;
#pragma unroll
  for (int nt = 0; nt < 4; ++nt) {
    float s = 0.f;
#pragma unroll
    for (int r = 0; r < 8; ++r) {
      float re = acc[nt][r], im = acc[nt+4][r];
      s += sqrtf(re*re + im*im);
    }
    atomicAdd(&fluxacc[t*64 + nt*16 + ln], s);
  }
}

// ---------------------------------------------------------------------------
// Flux / per-(t,d) operator tables.  Gate math cancels (g + (1-g) == 1), so
// Wg/bg never affect outputs; only flux_seq (for flux_next) is needed.
// ---------------------------------------------------------------------------
__global__ void k_flux(const float* __restrict__ fluxacc, const float* __restrict__ decay,
                       const float* __restrict__ flux_prev, const float* __restrict__ dt,
                       const float* __restrict__ lre, const float* __restrict__ lim,
                       float* __restrict__ Atab, float* __restrict__ Ftab,
                       float* __restrict__ out_flux) {
  int i = blockIdx.x*256 + threadIdx.x;
  if (i >= 1024) return;
  int t = i >> 6, d = i & 63;
  float mean  = fluxacc[i] * (1.0f/1024.0f);
  float alpha = 1.0f/(1.0f + expf(-decay[d]));
  float fs    = alpha*flux_prev[d] + (1.0f-alpha)*mean;
  if (t == 15) out_flux[d] = fs;
  float lr = lre[d];
  float sp = fmaxf(lr, 0.0f) + log1pf(expf(-fabsf(lr)));   // softplus
  float er = -sp, ei = lim[d];
  float dtv = dt[t];
  float mg  = expf(er*dtv);
  float exr = mg*cosf(ei*dtv), exi = mg*sinf(ei*dtv);
  Atab[2*i] = exr; Atab[2*i+1] = exi;
  float dr = er + 1e-8f, di = ei;
  float inv = 1.0f/(dr*dr + di*di);
  float nr = 1.0f - exr, ni = -exi;
  Ftab[2*i]   = (nr*dr + ni*di)*inv;
  Ftab[2*i+1] = (ni*dr - nr*di)*inv;
}

// ---------------------------------------------------------------------------
// Sequential scan over T (T=16): u[t] = A[t]*u[t-1] + xeig[t]*F[t]  (complex)
// ---------------------------------------------------------------------------
__global__ void k_scan(const float* __restrict__ xeig, const float* __restrict__ Atab,
                       const float* __restrict__ Ftab, _Float16* __restrict__ ucat,
                       float* __restrict__ out_h) {
  int i = blockIdx.x*256 + threadIdx.x;  // 65536
  int d = i & 63, hw = i >> 6;
  float ur = 0.f, ui = 0.f;
#pragma unroll
  for (int t = 0; t < 16; ++t) {
    int m = t*1024 + hw;
    float xr = xeig[(size_t)m*128 + d];
    float xi = xeig[(size_t)m*128 + 64 + d];
    int td = t*64 + d;
    float Fr = Ftab[2*td], Fi = Ftab[2*td+1];
    float Ar = Atab[2*td], Ai = Atab[2*td+1];
    float utr = xr*Fr - xi*Fi;
    float uti = xr*Fi + xi*Fr;
    float nr = Ar*ur - Ai*ui + utr;
    float ni = Ar*ui + Ai*ur + uti;
    ur = nr; ui = ni;
    ucat[(size_t)m*128 + d]      = (_Float16)ur;
    ucat[(size_t)m*128 + 64 + d] = (_Float16)ui;
  }
  out_h[hw*64 + d]         = ur;   // h_next real
  out_h[65536 + hw*64 + d] = ui;   // h_next imag
}

// ---------------------------------------------------------------------------
// Decoder GEMM (WMMA) + LayerNorm + router softmax
// ---------------------------------------------------------------------------
__global__ void __launch_bounds__(256) k_dec(const _Float16* __restrict__ ucat,
                                             const _Float16* __restrict__ WdecT,
                                             const float* __restrict__ Wr,
                                             const float* __restrict__ lng,
                                             const float* __restrict__ lnb,
                                             float* __restrict__ xc,
                                             _Float16* __restrict__ xc16,
                                             float* __restrict__ pbuf) {
  __shared__ __align__(16) _Float16 sW[128*128];
  __shared__ float sWr[128*8];
  __shared__ float sg[128], sb[128];
  __shared__ float stage[8][16][128];
  __shared__ float smu[8][16], srs[8][16];
  {
    const uint4* s = (const uint4*)WdecT;
    uint4* d = (uint4*)sW;
    for (int i = threadIdx.x; i < 128*128/8; i += 256) d[i] = s[i];
    for (int i = threadIdx.x; i < 1024; i += 256) sWr[i] = Wr[i];
    if (threadIdx.x < 128) { sg[threadIdx.x] = lng[threadIdx.x]; sb[threadIdx.x] = lnb[threadIdx.x]; }
  }
  __syncthreads();
  int lane = threadIdx.x & 31, wave = threadIdx.x >> 5;
  int g = lane >> 4, ln = lane & 15;
  int m0 = (blockIdx.x*8 + wave) * 16;

  v16h a[4];
#pragma unroll
  for (int kt = 0; kt < 4; ++kt) a[kt] = load_fragA(ucat + (size_t)(m0+ln)*128 + kt*32, g);

#pragma unroll
  for (int nt = 0; nt < 8; ++nt) {
    v8f c = {0.f,0.f,0.f,0.f,0.f,0.f,0.f,0.f};
#pragma unroll
    for (int kt = 0; kt < 4; ++kt) {
      v16h b = load_fragB(sW + (nt*16 + ln)*128 + kt*32, g);
      c = WMMA(a[kt], b, c);
    }
#pragma unroll
    for (int r = 0; r < 8; ++r) stage[wave][r + 8*g][nt*16 + ln] = c[r];
  }
  asm volatile("" ::: "memory");   // LDS is in-order per wave; block compiler reordering

  if (lane < 16) {
    float mu = 0.f;
    for (int k = 0; k < 128; ++k) mu += stage[wave][lane][k];
    mu *= (1.0f/128.0f);
    float var = 0.f;
    for (int k = 0; k < 128; ++k) { float dv = stage[wave][lane][k] - mu; var += dv*dv; }
    var *= (1.0f/128.0f);
    smu[wave][lane] = mu;
    srs[wave][lane] = rsqrtf(var + 1e-5f);
  }
  asm volatile("" ::: "memory");

  for (int idx = lane; idx < 16*128; idx += 32) {
    int m = idx >> 7, n = idx & 127;
    float v = (stage[wave][m][n] - smu[wave][m]) * srs[wave][m] * sg[n] + sb[n];
    stage[wave][m][n] = v;
    int gm = m0 + m;
    xc[(size_t)gm*128 + n]   = v;
    xc16[(size_t)gm*128 + n] = (_Float16)v;
  }
  asm volatile("" ::: "memory");

  if (lane < 16) {                      // router: p = softmax(x_norm @ Wr)
    int gm = m0 + lane;
    float z[8] = {0,0,0,0,0,0,0,0};
    for (int k = 0; k < 128; ++k) {
      float xv = stage[wave][lane][k];
#pragma unroll
      for (int e = 0; e < 8; ++e) z[e] += xv * sWr[k*8 + e];
    }
    float mx = z[0];
#pragma unroll
    for (int e = 1; e < 8; ++e) mx = fmaxf(mx, z[e]);
    float s = 0.f;
#pragma unroll
    for (int e = 0; e < 8; ++e) { z[e] = expf(z[e] - mx); s += z[e]; }
    float inv = 1.0f/s;
#pragma unroll
    for (int e = 0; e < 8; ++e) pbuf[gm*8 + e] = z[e]*inv;
  }
}

// ---------------------------------------------------------------------------
// 8-expert complex MLP (fused GEMM1 -> gelu -> GEMM2, per 32-col chunk),
// delta = sum_e p_e * y_e, then z = x + x_c + delta written transposed.
// ---------------------------------------------------------------------------
__global__ void __launch_bounds__(256) k_mlp(const _Float16* __restrict__ xc16,
                                             const _Float16* __restrict__ W1T,
                                             const _Float16* __restrict__ W2P,
                                             const float* __restrict__ b1cat,
                                             const float* __restrict__ b2cat,
                                             const float* __restrict__ pbuf,
                                             const float* __restrict__ xc,
                                             const float* __restrict__ xre,
                                             const float* __restrict__ xim,
                                             float* __restrict__ zout) {
  __shared__ __align__(16) _Float16 sW1[32*128];     // GEMM1 B slab (chunk of N)
  __shared__ __align__(16) _Float16 sW2[128*32];     // GEMM2 B slab (chunk of K)
  __shared__ __align__(16) _Float16 stage[8][16*32]; // per-wave gelu(H1) chunk

  int lane = threadIdx.x & 31, wave = threadIdx.x >> 5;
  int g = lane >> 4, ln = lane & 15;
  int m0 = (blockIdx.x*8 + wave) * 16;

  v16h a[4];
#pragma unroll
  for (int kt = 0; kt < 4; ++kt) a[kt] = load_fragA(xc16 + (size_t)(m0+ln)*128 + kt*32, g);

  v8f delta[8];
#pragma unroll
  for (int nt = 0; nt < 8; ++nt) delta[nt] = (v8f){0.f,0.f,0.f,0.f,0.f,0.f,0.f,0.f};

  for (int e = 0; e < 8; ++e) {
    v8f acc2[8];
#pragma unroll
    for (int nt = 0; nt < 8; ++nt) acc2[nt] = (v8f){0.f,0.f,0.f,0.f,0.f,0.f,0.f,0.f};

    for (int nc = 0; nc < 16; ++nc) {
      __syncthreads();
      {
        const uint4* s1 = (const uint4*)(W1T + ((size_t)e*512 + nc*32)*128);
        const uint4* s2 = (const uint4*)(W2P + ((size_t)(e*16 + nc)*128)*32);
        uint4* d1 = (uint4*)sW1;
        uint4* d2 = (uint4*)sW2;
        for (int i = threadIdx.x; i < 512; i += 256) { d1[i] = s1[i]; d2[i] = s2[i]; }
      }
      if (nc < 15 || e < 7) {          // prefetch next slabs into GL2
        int ne = (nc < 15) ? e : e + 1;
        int nn = (nc < 15) ? nc + 1 : 0;
        __builtin_prefetch(W1T + ((size_t)ne*512 + nn*32)*128, 0, 1);
        __builtin_prefetch(W2P + ((size_t)(ne*16 + nn)*128)*32, 0, 1);
      }
      __syncthreads();

      // GEMM1: H1 chunk (16 x 32) = X(16x128) @ W1[:, nc*32 .. +32)
      v8f acc1[2];
      acc1[0] = (v8f){0.f,0.f,0.f,0.f,0.f,0.f,0.f,0.f};
      acc1[1] = (v8f){0.f,0.f,0.f,0.f,0.f,0.f,0.f,0.f};
#pragma unroll
      for (int kt = 0; kt < 4; ++kt) {
        v16h b0 = load_fragB(sW1 + (0*16 + ln)*128 + kt*32, g);
        v16h b1 = load_fragB(sW1 + (1*16 + ln)*128 + kt*32, g);
        acc1[0] = WMMA(a[kt], b0, acc1[0]);
        acc1[1] = WMMA(a[kt], b1, acc1[1]);
      }
      // bias + gelu, stage as GEMM2 A fragment (C-layout -> A-layout transpose via LDS)
#pragma unroll
      for (int nt2 = 0; nt2 < 2; ++nt2) {
        float bv = b1cat[e*512 + nc*32 + nt2*16 + ln];
#pragma unroll
        for (int r = 0; r < 8; ++r) {
          float v = gelu_tanh(acc1[nt2][r] + bv);
          stage[wave][(r + 8*g)*32 + nt2*16 + ln] = (_Float16)v;
        }
      }
      asm volatile("" ::: "memory");   // wave-private LDS; in-order per wave
      v16h a2 = load_fragA(stage[wave] + ln*32, g);

      // GEMM2 partial: Y(16x128) += H1chunk(16x32) @ W2[nc*32 .. +32, :]
#pragma unroll
      for (int nt = 0; nt < 8; ++nt) {
        v16h b = load_fragB(sW2 + (nt*16 + ln)*32, g);
        acc2[nt] = WMMA(a2, b, acc2[nt]);
      }
    }

    // delta += p_e * (Y + b2)
    float b2v[8];
#pragma unroll
    for (int nt = 0; nt < 8; ++nt) b2v[nt] = b2cat[e*128 + nt*16 + ln];
#pragma unroll
    for (int r = 0; r < 8; ++r) {
      float pe = pbuf[(m0 + r + 8*g)*8 + e];
#pragma unroll
      for (int nt = 0; nt < 8; ++nt) delta[nt][r] += pe * (acc2[nt][r] + b2v[nt]);
    }
  }

  // z = x + (x_c + delta), transposed back to (2, T, D, H, W)
#pragma unroll
  for (int nt = 0; nt < 8; ++nt) {
#pragma unroll
    for (int r = 0; r < 8; ++r) {
      int m = m0 + r + 8*g;
      int n = nt*16 + ln;
      int t = m >> 10, hw = m & 1023;
      int d = n & 63,  s = n >> 6;
      float base = (s == 0) ? xre[(t*64 + d)*1024 + hw] : xim[(t*64 + d)*1024 + hw];
      float v = base + xc[(size_t)m*128 + n] + delta[nt][r];
      zout[((size_t)(s*16 + t)*64 + d)*1024 + hw] = v;
    }
  }
}

// ---------------------------------------------------------------------------
extern "C" void kernel_launch(void* const* d_in, const int* in_sizes, int n_in,
                              void* d_out, int out_size, void* d_ws, size_t ws_size,
                              hipStream_t stream) {
  const float* x_re      = (const float*)d_in[0];
  const float* x_im      = (const float*)d_in[1];
  const float* dt        = (const float*)d_in[2];
  const float* flux_prev = (const float*)d_in[3];
  const float* Uenc_re   = (const float*)d_in[4];
  const float* Uenc_im   = (const float*)d_in[5];
  const float* Udec_re   = (const float*)d_in[6];
  const float* Udec_im   = (const float*)d_in[7];
  const float* decay     = (const float*)d_in[8];
  // d_in[9] Wg, d_in[10] bg: unused — gate cancels algebraically
  const float* lre       = (const float*)d_in[11];
  const float* lim       = (const float*)d_in[12];
  const float* ln_g      = (const float*)d_in[13];
  const float* ln_b      = (const float*)d_in[14];
  const float* Wr        = (const float*)d_in[15];
  const float* W1_re     = (const float*)d_in[16];
  const float* W1_im     = (const float*)d_in[17];
  const float* b1_re     = (const float*)d_in[18];
  const float* b1_im     = (const float*)d_in[19];
  const float* W2_re     = (const float*)d_in[20];
  const float* W2_im     = (const float*)d_in[21];
  const float* b2_re     = (const float*)d_in[22];
  const float* b2_im     = (const float*)d_in[23];

  char* ws = (char*)d_ws;
  _Float16* WencT = (_Float16*)(ws + WS_WENC);
  _Float16* WdecT = (_Float16*)(ws + WS_WDEC);
  _Float16* W1T   = (_Float16*)(ws + WS_W1T);
  _Float16* W2P   = (_Float16*)(ws + WS_W2P);
  float*    b1cat = (float*)(ws + WS_B1);
  float*    b2cat = (float*)(ws + WS_B2);
  _Float16* Xcat  = (_Float16*)(ws + WS_XCAT);
  float*    xeig  = (float*)(ws + WS_XEIG);
  float*    fluxa = (float*)(ws + WS_FLUX);
  float*    Atab  = (float*)(ws + WS_ATAB);
  float*    Ftab  = (float*)(ws + WS_FTAB);
  _Float16* ucat  = (_Float16*)(ws + WS_UCAT);
  float*    xc    = (float*)(ws + WS_XC);
  _Float16* xc16  = (_Float16*)(ws + WS_XC16);
  float*    pbuf  = (float*)(ws + WS_P);

  float* zout = (float*)d_out;
  float* hout = (float*)d_out + OFF_H;
  float* fout = (float*)d_out + OFF_FLUX;

  k_pack_encdec<<<64,   256, 0, stream>>>(Uenc_re, Uenc_im, WencT);
  k_pack_encdec<<<64,   256, 0, stream>>>(Udec_re, Udec_im, WdecT);
  k_pack_w1    <<<2048, 256, 0, stream>>>(W1_re, W1_im, W1T);
  k_pack_w2    <<<2048, 256, 0, stream>>>(W2_re, W2_im, W2P);
  k_pack_misc  <<<24,   256, 0, stream>>>(b1_re, b1_im, b2_re, b2_im, b1cat, b2cat, fluxa);
  k_pack_x     <<<8192, 256, 0, stream>>>(x_re, x_im, Xcat);

  k_enc <<<128, 256, 0, stream>>>(Xcat, WencT, xeig, fluxa);
  k_flux<<<4,   256, 0, stream>>>(fluxa, decay, flux_prev, dt, lre, lim, Atab, Ftab, fout);
  k_scan<<<256, 256, 0, stream>>>(xeig, Atab, Ftab, ucat, hout);
  k_dec <<<128, 256, 0, stream>>>(ucat, WdecT, Wr, ln_g, ln_b, xc, xc16, pbuf);
  k_mlp <<<128, 256, 0, stream>>>(xc16, W1T, W2P, b1cat, b2cat, pbuf, xc, x_re, x_im, zout);

  (void)in_sizes; (void)n_in; (void)out_size; (void)ws_size;
}
